// GATv2Layer_7851200217926
// MI455X (gfx1250) — compile-verified, matrix operand order checked by reading, not run on hardware
//
#include <hip/hip_runtime.h>
#include <math.h>

// GAT layer on gfx1250 (MI455X): bf16 WMMA for both GEMMs, two-pass masked
// softmax with fully factorized exponentials (rank-1 logit structure),
// double-buffered WMMA pipelines. Zero transcendentals in hot loops.

typedef __attribute__((ext_vector_type(16))) __bf16   v16bf;
typedef __attribute__((ext_vector_type(8)))  float    v8f;
typedef __attribute__((ext_vector_type(8)))  unsigned v8u;

#define NN    4096
#define KD    512          // IN_DIM
#define OD    512          // OUT_DIM
#define NH    8
#define HD    64           // D
#define ALPHA 0.2f
#define NEGV  (-9.0e15f)

// Pack two f32 into a dword of two bf16 (lo, hi), round-to-nearest-even.
static __device__ __forceinline__ unsigned pk_bf16(float lo, float hi) {
#if __has_builtin(__builtin_amdgcn_cvt_pk_bf16_f32)
  return __builtin_bit_cast(unsigned, __builtin_amdgcn_cvt_pk_bf16_f32(lo, hi));
#else
  unsigned ul = __builtin_bit_cast(unsigned, lo);
  unsigned uh = __builtin_bit_cast(unsigned, hi);
  ul += 0x7FFFu + ((ul >> 16) & 1u);
  uh += 0x7FFFu + ((uh >> 16) & 1u);
  return (ul >> 16) | (uh & 0xFFFF0000u);
#endif
}

static __device__ __forceinline__ unsigned short f2bfs(float f) {
  return (unsigned short)(pk_bf16(f, f) & 0xFFFFu);
}

static __device__ __forceinline__ int kbase(int p, int half) {
  // ISA 16-bit A/B fragment K mapping for 16x16x32 (wave32)
  return (p < 4 ? (p << 1) : (16 + ((p - 4) << 1))) + (half << 3);
}

static __device__ __forceinline__ float lrelu(float v) {
  return v > 0.f ? v : ALPHA * v;
}

// ---------------------------------------------------------------------------
// Kernel 0: elementwise f32 -> packed bf16 (two elements per thread)
// ---------------------------------------------------------------------------
__global__ __launch_bounds__(256) void cvt_bf16(const float* __restrict__ in,
                                                unsigned* __restrict__ out) {
  const size_t t = (size_t)blockIdx.x * 256 + threadIdx.x;
  const float2 v = *(const float2*)(in + 2 * t);
  out[t] = pk_bf16(v.x, v.y);
}

// ---------------------------------------------------------------------------
// Kernel 1: Wh = x @ W^T  (bf16 inputs, f32 accum), each wave: 16x64 tile.
// Double-buffered over K.
// ---------------------------------------------------------------------------
__global__ __launch_bounds__(256) void wh_gemm(const unsigned short* __restrict__ xbf,
                                               const unsigned short* __restrict__ Wbf,
                                               float* __restrict__ Wh) {
  const int wave = threadIdx.x >> 5;
  const int lane = threadIdx.x & 31;
  const int half = lane >> 4;
  const int idx  = lane & 15;
  const int gw   = blockIdx.x * 8 + wave;     // 0..2047
  const int rowBase = (gw >> 3) * 16;         // 256 row tiles
  const int colBase = (gw & 7) * 64;          // 8 col tiles of 64

  v8f zero = {};
  v8f acc[4] = {zero, zero, zero, zero};
  v8u aU[2];
  v8u bU[2][4];

#define G_LOAD(st, k0v) do {                                                  \
    const int k0_ = (k0v);                                                    \
    const unsigned short* xp = xbf + (size_t)(rowBase + idx) * KD + k0_;      \
    _Pragma("unroll")                                                         \
    for (int p = 0; p < 8; ++p)                                               \
      aU[st][p] = *(const unsigned*)(xp + kbase(p, half));                    \
    _Pragma("unroll")                                                         \
    for (int t = 0; t < 4; ++t) {                                             \
      const unsigned short* wp =                                              \
          Wbf + (size_t)(colBase + t * 16 + idx) * KD + k0_;                  \
      _Pragma("unroll")                                                       \
      for (int p = 0; p < 8; ++p)                                             \
        bU[st][t][p] = *(const unsigned*)(wp + kbase(p, half));               \
    }                                                                         \
  } while (0)

#define G_MMA(st) do {                                                        \
    const v16bf af = __builtin_bit_cast(v16bf, aU[st]);                       \
    _Pragma("unroll")                                                         \
    for (int t = 0; t < 4; ++t) {                                             \
      const v16bf bf_ = __builtin_bit_cast(v16bf, bU[st][t]);                 \
      acc[t] = __builtin_amdgcn_wmma_f32_16x16x32_bf16(                       \
          false, af, false, bf_, (short)0, acc[t], false, false);             \
    }                                                                         \
  } while (0)

  G_LOAD(0, 0);
  for (int k0 = 0; k0 < KD; k0 += 64) {
    G_LOAD(1, k0 + 32);
    G_MMA(0);
    if (k0 + 64 < KD) G_LOAD(0, k0 + 64);
    G_MMA(1);
  }
#undef G_LOAD
#undef G_MMA

#pragma unroll
  for (int t = 0; t < 4; ++t)
#pragma unroll
    for (int r = 0; r < 8; ++r)
      Wh[(size_t)(rowBase + r + 8 * half) * OD + colBase + t * 16 + idx] = acc[t][r];
}

// ---------------------------------------------------------------------------
// Kernel 1b: WhbfT[c][j] = bf16(Wh[j][c])  — LDS-tiled transpose + convert.
// ---------------------------------------------------------------------------
__global__ __launch_bounds__(256) void wh_transpose_bf(const float* __restrict__ Wh,
                                                       unsigned short* __restrict__ WhbfT) {
  __shared__ float tile[32][33];
  const int jb = (blockIdx.x & 127) * 32;
  const int cb = (blockIdx.x >> 7) * 32;
  const int tr = threadIdx.x >> 5;            // 0..7
  const int tc = threadIdx.x & 31;
#pragma unroll
  for (int k = 0; k < 4; ++k) {
    const int r = k * 8 + tr;
    tile[r][tc] = Wh[(size_t)(jb + r) * OD + cb + tc];
  }
  __syncthreads();
#pragma unroll
  for (int k = 0; k < 4; ++k) {
    const int r = k * 8 + tr;                 // c offset within tile
    WhbfT[(size_t)(cb + r) * NN + jb + tc] = f2bfs(tile[tc][r]);
  }
}

// ---------------------------------------------------------------------------
// Kernel 2: per (j,h): ss = Wh[j,h,:].a[0:64], sd = Wh[j,h,:].a[64:128]
// Also emits EPT[h][j] = (exp(ss), exp(ALPHA*ss)) for exp factorization.
// ---------------------------------------------------------------------------
__global__ __launch_bounds__(256) void s_proj(const float* __restrict__ Wh,
                                              const float* __restrict__ a,
                                              float* __restrict__ SsrcT,
                                              float2* __restrict__ EPT,
                                              float* __restrict__ Sdst) {
  const int t = blockIdx.x * 256 + threadIdx.x;    // N*H threads
  const int j = t >> 3, h = t & 7;
  const float* w = Wh + (size_t)j * OD + h * HD;
  float s0 = 0.f, s1 = 0.f;
#pragma unroll 8
  for (int d = 0; d < HD; ++d) {
    const float wv = w[d];
    s0 += wv * a[d];
    s1 += wv * a[HD + d];
  }
  SsrcT[(size_t)h * NN + j] = s0;
  EPT[(size_t)h * NN + j] = make_float2(__expf(s0), __expf(ALPHA * s0));
  Sdst[t] = s1;
}

// ---------------------------------------------------------------------------
// Kernel 3: per (i,h): m = max_j e[i,j,h], l = sum_j exp(e - m).
// Pass 1: masked max of s_src (lrelu is monotone). Pass 2: factorized
// exponentials — zero exps in the loop. One block per row; one wave per head.
// ---------------------------------------------------------------------------
__global__ __launch_bounds__(256) void softmax_stats(const int* __restrict__ adj,
                                                     const float* __restrict__ SsrcT,
                                                     const float2* __restrict__ EPT,
                                                     const float* __restrict__ Sdst,
                                                     float* __restrict__ Mrow,
                                                     float* __restrict__ Lrec) {
  __shared__ unsigned char adjb[NN];
  const int i = blockIdx.x;
  for (int j = threadIdx.x; j < NN; j += 256)
    adjb[j] = (adj[(size_t)i * NN + j] > 0) ? 1 : 0;
  __syncthreads();

  const int h    = threadIdx.x >> 5;     // wave -> head
  const int lane = threadIdx.x & 31;
  const float sd = Sdst[i * NH + h];
  const float*  ss = SsrcT + (size_t)h * NN;
  const float2* ep = EPT + (size_t)h * NN;

  // pass 1: masked max of s_src[j]
  float mss = -INFINITY;
  for (int j = lane; j < NN; j += 32)
    if (adjb[j]) mss = fmaxf(mss, ss[j]);
#pragma unroll
  for (int off = 16; off >= 1; off >>= 1)
    mss = fmaxf(mss, __shfl_xor(mss, off, 32));

  float m, s;
  if (mss == -INFINITY) {          // row has no neighbors: e == NEG everywhere
    m = NEGV;
    s = (float)NN;                 // every exp(e-m) == 1
  } else {
    m = lrelu(sd + mss);
    const float c1 = __expf(sd - m);            // pos branch row factor
    const float c2 = __expf(ALPHA * sd - m);    // neg branch row factor
    float ps = 0.f;
    for (int j = lane; j < NN; j += 32)
      if (adjb[j]) {
        const float  sj = ss[j];
        const float2 ej = ep[j];
        ps += (sd + sj > 0.f) ? c1 * ej.x : c2 * ej.y;
      }
#pragma unroll
    for (int off = 16; off >= 1; off >>= 1)
      ps += __shfl_xor(ps, off, 32);
    s = ps;
  }
  if (lane == 0) {
    Mrow[i * NH + h] = m;
    Lrec[i * NH + h] = 1.0f / s;
  }
}

// ---------------------------------------------------------------------------
// Kernel 4: h'[i,h,:] = (1/l) * sum_j P[i,j] * Wh[j,h,:]  via WMMA.
// P built with factorized exponentials (no transcendentals in loop).
// Each wave: 32 rows (two A tiles sharing B) x one head x 64 cols.
// ---------------------------------------------------------------------------
__global__ __launch_bounds__(256) void attn_wmma(const int* __restrict__ adj,
                                                 const unsigned short* __restrict__ WhbfT,
                                                 const float* __restrict__ SsrcT,
                                                 const float2* __restrict__ EPT,
                                                 const float* __restrict__ Sdst,
                                                 const float* __restrict__ Mrow,
                                                 const float* __restrict__ Lrec,
                                                 float* __restrict__ HP) {
  const int wave = threadIdx.x >> 5;
  const int lane = threadIdx.x & 31;
  const int half = lane >> 4;
  const int idx  = lane & 15;
  const int gw   = blockIdx.x * 8 + wave;     // 0..1023
  const int iBase = (gw >> 3) * 32;           // 128 tiles of 32 rows
  const int h     = gw & 7;                   // head
  const int i0    = iBase + idx;              // A-tile-0 row for this lane
  const int i1    = iBase + 16 + idx;         // A-tile-1 row for this lane

  const float sd0 = Sdst[i0 * NH + h], mi0 = Mrow[i0 * NH + h];
  const float sd1 = Sdst[i1 * NH + h], mi1 = Mrow[i1 * NH + h];
  // factorized row constants: exp(e-m) = c1*exp(ss) (pos) or c2*exp(a*ss) (neg)
  const float c10 = __expf(sd0 - mi0), c20 = __expf(ALPHA * sd0 - mi0);
  const float c11 = __expf(sd1 - mi1), c21 = __expf(ALPHA * sd1 - mi1);
  const float pz0 = (mi0 == NEGV) ? 1.f : 0.f;   // uniform fallback, empty rows
  const float pz1 = (mi1 == NEGV) ? 1.f : 0.f;
  const size_t ar0 = (size_t)i0 * NN;
  const size_t ar1 = (size_t)i1 * NN;
  const float* ss = SsrcT + (size_t)h * NN;
  const float* ep = (const float*)(EPT + (size_t)h * NN);

  const unsigned short* wcol[4];
#pragma unroll
  for (int t = 0; t < 4; ++t)
    wcol[t] = WhbfT + (size_t)(h * HD + t * 16 + idx) * NN;

  v8f zero = {};
  v8f acc[8] = {zero, zero, zero, zero, zero, zero, zero, zero};
  v8u aU0[2], aU1[2];
  v8u bU[2][4];

#define A_LOAD(st, j0v) do {                                                  \
    const int jb_ = (j0v);                                                    \
    __builtin_prefetch(adj + ar0 + jb_ + 512, 0, 0);                          \
    __builtin_prefetch(adj + ar1 + jb_ + 512, 0, 0);                          \
    _Pragma("unroll")                                                         \
    for (int p = 0; p < 8; ++p) {                                             \
      const int kb = kbase(p, half);                                          \
      const int j  = jb_ + kb;                                                \
      const int2   a0 = *(const int2*)(adj + ar0 + j);                        \
      const int2   a1 = *(const int2*)(adj + ar1 + j);                        \
      const float2 sv = *(const float2*)(ss + j);                             \
      const float4 ej = *(const float4*)(ep + 2 * j);                         \
      const float p00 = (a0.x > 0)                                            \
          ? ((sd0 + sv.x > 0.f) ? c10 * ej.x : c20 * ej.y) : pz0;             \
      const float p01 = (a0.y > 0)                                            \
          ? ((sd0 + sv.y > 0.f) ? c10 * ej.z : c20 * ej.w) : pz0;             \
      const float p10 = (a1.x > 0)                                            \
          ? ((sd1 + sv.x > 0.f) ? c11 * ej.x : c21 * ej.y) : pz1;             \
      const float p11 = (a1.y > 0)                                            \
          ? ((sd1 + sv.y > 0.f) ? c11 * ej.z : c21 * ej.w) : pz1;             \
      aU0[st][p] = pk_bf16(p00, p01);                                         \
      aU1[st][p] = pk_bf16(p10, p11);                                         \
    }                                                                         \
    _Pragma("unroll")                                                         \
    for (int t = 0; t < 4; ++t) {                                             \
      const unsigned short* wp = wcol[t] + jb_;                               \
      _Pragma("unroll")                                                       \
      for (int p = 0; p < 8; ++p)                                             \
        bU[st][t][p] = *(const unsigned*)(wp + kbase(p, half));               \
    }                                                                         \
  } while (0)

#define A_MMA(st) do {                                                        \
    const v16bf a0f = __builtin_bit_cast(v16bf, aU0[st]);                     \
    const v16bf a1f = __builtin_bit_cast(v16bf, aU1[st]);                     \
    _Pragma("unroll")                                                         \
    for (int t = 0; t < 4; ++t) {                                             \
      const v16bf bf_ = __builtin_bit_cast(v16bf, bU[st][t]);                 \
      acc[t]     = __builtin_amdgcn_wmma_f32_16x16x32_bf16(                   \
          false, a0f, false, bf_, (short)0, acc[t],     false, false);        \
      acc[4 + t] = __builtin_amdgcn_wmma_f32_16x16x32_bf16(                   \
          false, a1f, false, bf_, (short)0, acc[4 + t], false, false);        \
    }                                                                         \
  } while (0)

  A_LOAD(0, 0);
  for (int j0 = 0; j0 < NN; j0 += 64) {
    A_LOAD(1, j0 + 32);
    A_MMA(0);
    if (j0 + 64 < NN) A_LOAD(0, j0 + 64);
    A_MMA(1);
  }
#undef A_LOAD
#undef A_MMA

  float linv0[8], linv1[8];
#pragma unroll
  for (int r = 0; r < 8; ++r) {
    linv0[r] = Lrec[(iBase + r + 8 * half) * NH + h];
    linv1[r] = Lrec[(iBase + 16 + r + 8 * half) * NH + h];
  }
#pragma unroll
  for (int t = 0; t < 4; ++t)
#pragma unroll
    for (int r = 0; r < 8; ++r) {
      HP[(size_t)(iBase + r + 8 * half) * OD + h * HD + t * 16 + idx] =
          acc[t][r] * linv0[r];
      HP[(size_t)(iBase + 16 + r + 8 * half) * OD + h * HD + t * 16 + idx] =
          acc[4 + t][r] * linv1[r];
    }
}

// ---------------------------------------------------------------------------
// Kernel 5: per-feature mean / rstd over the 4096 rows.
// ---------------------------------------------------------------------------
__global__ __launch_bounds__(256) void col_stats(const float* __restrict__ HP,
                                                 float* __restrict__ Mean,
                                                 float* __restrict__ Rstd) {
  __shared__ float ss[256], sq[256];
  const int c = blockIdx.x;
  float s = 0.f, q = 0.f;
  for (int r = threadIdx.x; r < NN; r += 256) {
    const float v = HP[(size_t)r * OD + c];
    s += v;
    q += v * v;
  }
  ss[threadIdx.x] = s;
  sq[threadIdx.x] = q;
  __syncthreads();
  for (int o = 128; o >= 1; o >>= 1) {
    if (threadIdx.x < o) {
      ss[threadIdx.x] += ss[threadIdx.x + o];
      sq[threadIdx.x] += sq[threadIdx.x + o];
    }
    __syncthreads();
  }
  if (threadIdx.x == 0) {
    const float mean = ss[0] * (1.0f / NN);
    const float var  = sq[0] * (1.0f / NN) - mean * mean;
    Mean[c] = mean;
    Rstd[c] = rsqrtf(var + 1e-5f);
  }
}

// ---------------------------------------------------------------------------
// Kernel 6: out = relu(gamma*(h'-mean)*rstd + beta)
// ---------------------------------------------------------------------------
__global__ __launch_bounds__(256) void norm_relu(const float* __restrict__ HP,
                                                 const float* __restrict__ gamma,
                                                 const float* __restrict__ beta,
                                                 const float* __restrict__ Mean,
                                                 const float* __restrict__ Rstd,
                                                 float* __restrict__ out) {
  const size_t t = (size_t)blockIdx.x * 256 + threadIdx.x;
  const int c = (int)(t & (OD - 1));
  const float v = gamma[c] * (HP[t] - Mean[c]) * Rstd[c] + beta[c];
  out[t] = fmaxf(v, 0.f);
}

// ---------------------------------------------------------------------------
extern "C" void kernel_launch(void* const* d_in, const int* in_sizes, int n_in,
                              void* d_out, int out_size, void* d_ws, size_t ws_size,
                              hipStream_t stream) {
  const float* x     = (const float*)d_in[0];
  const int*   adj   = (const int*)d_in[1];
  const float* W     = (const float*)d_in[2];
  const float* a     = (const float*)d_in[3];
  const float* gamma = (const float*)d_in[4];
  const float* beta  = (const float*)d_in[5];

  float* ws = (float*)d_ws;
  float* Wh    = ws;                          // 4096*512 f32
  float* HP    = Wh + (size_t)NN * OD;        // 4096*512 f32
  float* SsrcT = HP + (size_t)NN * OD;        // 8*4096 f32 (head-major)
  float* Sdst  = SsrcT + NN * NH;             // 4096*8 f32
  float* Mrow  = Sdst + NN * NH;
  float* Lrec  = Mrow + NN * NH;
  float2* EPT  = (float2*)(Lrec + NN * NH);   // 8*4096 float2 (16B-aligned)
  float* Mean  = (float*)(EPT + (size_t)NN * NH);   // 512
  float* Rstd  = Mean + OD;                   // 512
  unsigned short* xbf   = (unsigned short*)(Rstd + OD);       // 4096*512 bf16
  unsigned short* Wbf   = xbf + (size_t)NN * KD;              // 512*512  bf16
  unsigned short* WhbfT = Wbf + (size_t)OD * KD;              // 512*4096 bf16

  cvt_bf16<<<(NN * KD / 2) / 256, 256, 0, stream>>>(x, (unsigned*)xbf);
  cvt_bf16<<<(OD * KD / 2) / 256, 256, 0, stream>>>(W, (unsigned*)Wbf);
  wh_gemm<<<256, 256, 0, stream>>>(xbf, Wbf, Wh);
  wh_transpose_bf<<<(NN / 32) * (OD / 32), 256, 0, stream>>>(Wh, WhbfT);
  s_proj<<<(NN * NH) / 256, 256, 0, stream>>>(Wh, a, SsrcT, EPT, Sdst);
  softmax_stats<<<NN, 256, 0, stream>>>(adj, SsrcT, EPT, Sdst, Mrow, Lrec);
  attn_wmma<<<128, 256, 0, stream>>>(adj, WhbfT, SsrcT, EPT, Sdst, Mrow, Lrec, HP);
  col_stats<<<OD, 256, 0, stream>>>(HP, Mean, Rstd);
  norm_relu<<<(NN * OD) / 256, 256, 0, stream>>>(HP, gamma, beta, Mean, Rstd,
                                                 (float*)d_out);
}
